// RealNetLM_6012954214631
// MI455X (gfx1250) — matrix operand order repaired and checked by reading, not compiled, not run on hardware
//
#include <hip/hip_runtime.h>
#include <math.h>

typedef __bf16 bf16;
typedef __attribute__((ext_vector_type(16))) __bf16 v16bf;
typedef __attribute__((ext_vector_type(8)))  __bf16 v8bf;
typedef __attribute__((ext_vector_type(8)))  float  v8f;

#define DEV __device__ __forceinline__

#define VSZ   32000
#define EDIM  768
#define NDIM  1024
#define BDIM  8
#define TDIM  256
#define THINK 4
#define MROWS (BDIM*TDIM)   // 2048
#define RWGS   16           // cooperative workgroups in recurrent kernel
#define RWAVES 4            // waves per workgroup (64 waves total = 64 N-tiles)
// TDM-padded LDS row stride: 1024 bf16 data + 4 dwords pad per 256-dword
// interval (2 intervals/row) -> 1024 + 16 bf16 = 1040
#define LDSK  1040

#if __has_builtin(__builtin_amdgcn_tensor_load_to_lds)
#define HAVE_TDM 1
typedef __attribute__((ext_vector_type(4))) unsigned u32x4;
typedef __attribute__((ext_vector_type(8))) int      i32x8;
typedef __attribute__((ext_vector_type(4))) int      i32x4;
#else
#define HAVE_TDM 0
#endif

// ---- WMMA helpers (gfx1250 16x16x32 bf16, f32 accum) ----------------------

DEV v8f wmma_bf16(v16bf a, v16bf b, v8f c) {
  // (neg_a, A, neg_b, B, c_mod, C, reuse_a, reuse_b)
  return __builtin_amdgcn_wmma_f32_16x16x32_bf16(false, a, false, b, (short)0, c,
                                                 false, false);
}

// A fragment (16x32, 16-bit): lane holds row M=lane&15; K = klo..klo+7 and
// klo+16..klo+23 with klo=(lane>>4)*8.  p must already include klo.
DEV v16bf fragA(const bf16* p) {
  v8bf lo = *(const v8bf*)(p);
  v8bf hi = *(const v8bf*)(p + 16);
  v16bf r;
#pragma unroll
  for (int i = 0; i < 8; ++i) { r[i] = lo[i]; r[i + 8] = hi[i]; }
  return r;
}

// B fragment (32x16, 16-bit): lane holds col N=lane&15; K = kb..kb+7 and
// kb+16..kb+23 with kb=(lane>>4)*8.  p must already include kb.  (B^T stored N-major, K contig.)
DEV v16bf fragB(const bf16* p) {
  v8bf lo = *(const v8bf*)(p);
  v8bf hi = *(const v8bf*)(p + 16);
  v16bf r;
#pragma unroll
  for (int i = 0; i < 8; ++i) { r[i] = lo[i]; r[i + 8] = hi[i]; }
  return r;
}

// ---- grid-wide barrier (cooperative, monotonic ticket) ---------------------

DEV void grid_sync(unsigned* bar) {
  __syncthreads();
  if (threadIdx.x == 0) {
    __threadfence();                      // release prior writes
    unsigned ticket = atomicAdd(bar, 1u);
    unsigned target = (ticket / RWGS + 1u) * RWGS;
    while (__hip_atomic_load(bar, __ATOMIC_RELAXED, __HIP_MEMORY_SCOPE_AGENT) < target) { }
  }
#if __has_builtin(__builtin_amdgcn_s_cluster_barrier)
  __builtin_amdgcn_s_cluster_barrier();  // s_barrier_signal/wait -3; NOP when not a cluster
#endif
  __syncthreads();
  __threadfence();                        // acquire: invalidate local caches in all threads
}

// ---- weight convert+transpose: out[j][k] = (bf16) in[k][j] -----------------

__global__ void k_convert_T(const float* __restrict__ in, bf16* __restrict__ out,
                            int K, int J) {
  long tid = (long)blockIdx.x * blockDim.x + threadIdx.x;
  long total = (long)K * J;
  if (tid >= total) return;
  int k = (int)(tid % K);
  int j = (int)(tid / K);
  out[(long)j * K + k] = (bf16)in[(long)k * J + j];
}

// ---- x = emb[ids] @ w_in + b_in : gathered-A bf16 WMMA, 16x64 strip/wave ---

__global__ void k_embed_proj(const int* __restrict__ ids, const float* __restrict__ emb,
                             const bf16* __restrict__ w_inT, const float* __restrict__ b_in,
                             float* __restrict__ x) {
  const int lane = threadIdx.x & 31;
  const int wv   = threadIdx.x >> 5;
  const int g    = blockIdx.x * (blockDim.x >> 5) + wv;
  const int Ng = NDIM >> 6;                 // groups of 64 output cols
  const int ng = g % Ng, mt = g / Ng;
  if (mt >= (MROWS >> 4)) return;
  const int cl    = lane & 15;
  const int klo   = (lane >> 4) * 8;
  const int rbase = (lane >> 4) * 8;

  const float* arow = emb + (size_t)ids[mt * 16 + cl] * EDIM;
  const bf16*  bp   = w_inT + (size_t)(ng * 64 + cl) * EDIM + ((lane >> 4) << 3);

  v8f acc[4];
#pragma unroll
  for (int j = 0; j < 4; ++j) {
    const float bias = b_in[ng * 64 + 16 * j + cl];
#pragma unroll
    for (int i = 0; i < 8; ++i) acc[j][i] = bias;
  }

  for (int k0 = 0; k0 < EDIM; k0 += 32) {
    const float* apf = arow + k0 + klo;
    float4 f0 = *(const float4*)(apf);
    float4 f1 = *(const float4*)(apf + 4);
    float4 f2 = *(const float4*)(apf + 16);
    float4 f3 = *(const float4*)(apf + 20);
    v16bf a;
    a[0]=(bf16)f0.x;  a[1]=(bf16)f0.y;  a[2]=(bf16)f0.z;  a[3]=(bf16)f0.w;
    a[4]=(bf16)f1.x;  a[5]=(bf16)f1.y;  a[6]=(bf16)f1.z;  a[7]=(bf16)f1.w;
    a[8]=(bf16)f2.x;  a[9]=(bf16)f2.y;  a[10]=(bf16)f2.z; a[11]=(bf16)f2.w;
    a[12]=(bf16)f3.x; a[13]=(bf16)f3.y; a[14]=(bf16)f3.z; a[15]=(bf16)f3.w;
#pragma unroll
    for (int j = 0; j < 4; ++j)
      acc[j] = wmma_bf16(a, fragB(bp + (size_t)(16 * j) * EDIM + k0), acc[j]);
  }
#pragma unroll
  for (int j = 0; j < 4; ++j)
#pragma unroll
    for (int i = 0; i < 8; ++i)
      x[(size_t)(mt * 16 + rbase + i) * NDIM + ng * 64 + 16 * j + cl] = acc[j][i];
}

// ---- cooperative recurrent core: 1024 dependent (16x1024)x(1024x1024) ------

__global__ void k_recurrent(const float* __restrict__ x, const bf16* __restrict__ w_bbT,
                            const float* __restrict__ b_bb, const float* __restrict__ gamma,
                            const float* __restrict__ beta,
                            bf16* __restrict__ a_st,   // 16 x NDIM bf16 (rows 8..15 zero)
                            float* __restrict__ s_st,  // 8 x NDIM f32 carry
                            bf16* __restrict__ h,      // (B*T) x NDIM bf16 layernorm out
                            unsigned* __restrict__ bar) {
  extern __shared__ bf16 lds[];
  const int lane = threadIdx.x & 31;
  const int wv   = threadIdx.x >> 5;
  const int g    = blockIdx.x * RWAVES + wv;      // 0..63 -> N-tile
  const int n0   = g * 16;
  const int cl   = lane & 15;
  const int klo  = (lane >> 4) * 8;
  bf16* myl = lds + (size_t)wv * 16 * LDSK;

  // Stage this wave's 16-column slice of w_bb^T into LDS once (reused 1024
  // steps).  LDS layout: 16 rows, padded stride LDSK=1040 bf16 with 8 pad
  // elems inserted after every 512 data elems (matches TDM pad engine).
#if HAVE_TDM
  {
    const unsigned lds_byte = (unsigned)(unsigned long long)(const void*)myl;
    const unsigned long long ga =
        (unsigned long long)(const void*)(w_bbT + (size_t)n0 * NDIM);
    u32x4 g0;
    g0[0] = 1u;                                   // count=1, user descriptor
    g0[1] = lds_byte;                             // lds_addr
    g0[2] = (unsigned)ga;                         // global_addr[31:0]
    g0[3] = (unsigned)((ga >> 32) & 0x1FFFFFFu) | (2u << 30);  // ga[56:32] | type=2
    i32x8 g1;
    g1[0] = (1 << 16)      // data_size = 2 bytes
          | (1 << 20)      // pad_enable
          | (7 << 22)      // pad_interval: 256 dwords
          | (3 << 25);     // pad_amount: 4 dwords (16B -> keeps alignment)
    g1[1] = (int)(NDIM << 16);            // tensor_dim0 low16 (bits 63:48)
    g1[2] = (int)(16 << 16);              // tensor_dim0 hi16=0 | tensor_dim1 lo16=16
    g1[3] = (int)(NDIM << 16);            // tensor_dim1 hi16=0 | tile_dim0=1024
    g1[4] = 16;                           // tile_dim1=16, tile_dim2=0
    g1[5] = NDIM;                         // tensor_dim0_stride low32
    g1[6] = 0;                            // stride hi | tensor_dim1_stride lo
    g1[7] = 0;
    i32x4 z4 = {0, 0, 0, 0};
#if __clang_major__ >= 23
    i32x8 z8 = {0, 0, 0, 0, 0, 0, 0, 0};
    __builtin_amdgcn_tensor_load_to_lds(g0, g1, z4, z4, z8, 0);
#else
    __builtin_amdgcn_tensor_load_to_lds(g0, g1, z4, z4, 0);
#endif
  }
#if __has_builtin(__builtin_amdgcn_s_wait_tensorcnt)
  __builtin_amdgcn_s_wait_tensorcnt(0);
#endif
#else
  for (int r = 0; r < 16; ++r) {
    const bf16* src = w_bbT + (size_t)(n0 + r) * NDIM;
    bf16* dst = myl + r * LDSK;
    for (int c = lane * 8; c < NDIM; c += 32 * 8)
      *(v8bf*)(dst + c + 8 * (c >> 9)) = *(const v8bf*)(src + c);
  }
#endif
  __syncthreads();

  if (blockIdx.x == 0) {
    for (int i = threadIdx.x; i < 8 * NDIM; i += blockDim.x) {
      s_st[i] = 0.f;                       // backbone.reset_state
      a_st[8 * NDIM + i] = (bf16)0.f;      // zero pad rows 8..15 of A
    }
  }
  grid_sync(bar);

  const float bias = b_bb[n0 + cl];
  const int tidg = blockIdx.x * blockDim.x + threadIdx.x;
  const int nthreads = RWGS * (RWAVES * 32);
  const bf16* ap = a_st + (size_t)cl * NDIM + klo;
  const bf16* bp = myl + (size_t)cl * LDSK + ((lane >> 4) << 3);

  for (int t = 0; t < TDIM; ++t) {
    // pulse: A = bf16(state + x_t)
    for (int i = tidg; i < 8 * NDIM; i += nthreads) {
      int b = i >> 10, n = i & (NDIM - 1);
      a_st[i] = (bf16)(s_st[i] + x[((size_t)b * TDIM + t) * NDIM + n]);
    }
    grid_sync(bar);

    for (int th = 0; th < THINK; ++th) {
      v8f acc;
#pragma unroll
      for (int i = 0; i < 8; ++i) acc[i] = bias;
      for (int k0 = 0; k0 < NDIM; k0 += 32) {
        v16bf a = fragA(ap + k0);
        v16bf b = fragB(bp + k0 + 8 * (k0 >> 9));   // skip TDM pad elems
        acc = wmma_bf16(a, b, acc);
      }
      if (lane < 16) {  // rows 0..7 valid (lanes 0-15 hold M=0..7)
        if (th < THINK - 1) {
#pragma unroll
          for (int i = 0; i < 8; ++i)
            a_st[(size_t)i * NDIM + n0 + cl] = (bf16)tanhf(acc[i]);
        } else {
#pragma unroll
          for (int i = 0; i < 8; ++i)
            s_st[(size_t)i * NDIM + n0 + cl] = tanhf(acc[i]);
        }
      }
      grid_sync(bar);
    }

    // layernorm of final s: global waves 0..7 handle batch rows 0..7
    if (g < 8) {
      float vals[NDIM / 32];
      float sum = 0.f, sq = 0.f;
#pragma unroll
      for (int j = 0; j < NDIM / 32; ++j) {
        float v = s_st[(size_t)g * NDIM + lane + 32 * j];
        vals[j] = v; sum += v; sq += v * v;
      }
#pragma unroll
      for (int off = 16; off >= 1; off >>= 1) {
        sum += __shfl_xor(sum, off, 32);
        sq  += __shfl_xor(sq,  off, 32);
      }
      const float mean = sum * (1.f / NDIM);
      const float var  = sq * (1.f / NDIM) - mean * mean;
      const float rs   = rsqrtf(var + 1e-5f);
#pragma unroll
      for (int j = 0; j < NDIM / 32; ++j) {
        int n = lane + 32 * j;
        float hv = (vals[j] - mean) * rs * gamma[n] + beta[n];
        h[((size_t)g * TDIM + t) * NDIM + n] = (bf16)hv;
      }
    }
  }
}

// ---- bf16 WMMA GEMM, 16(M)x64(N) strip/wave, software-pipelined K ----------

__global__ void k_gemm(const bf16* __restrict__ A, const bf16* __restrict__ BT,
                       float* __restrict__ Cf, bf16* __restrict__ Cb,
                       int M, int N, int K) {
  const int lane = threadIdx.x & 31;
  const int wv   = threadIdx.x >> 5;
  const int g    = blockIdx.x * (blockDim.x >> 5) + wv;
  const int Ng = N >> 6;
  const int ng = g % Ng, mt = g / Ng;
  if (mt >= (M >> 4)) return;
  const int cl    = lane & 15;
  const int klo   = (lane >> 4) * 8;
  const int rbase = (lane >> 4) * 8;

  const bf16* ap = A  + (size_t)(mt * 16 + cl) * K + klo;
  const bf16* bp = BT + (size_t)(ng * 64 + cl) * K + ((lane >> 4) << 3);

  v8f acc[4];
#pragma unroll
  for (int j = 0; j < 4; ++j)
#pragma unroll
    for (int i = 0; i < 8; ++i) acc[j][i] = 0.f;

  // software pipeline: loads for step s+1 issued before WMMAs of step s.
  // Tail prefetch reads past the row end stay inside d_ws (harmless).
  v16bf a = fragA(ap);
  v16bf b0 = fragB(bp);
  v16bf b1 = fragB(bp + (size_t)16 * K);
  v16bf b2 = fragB(bp + (size_t)32 * K);
  v16bf b3 = fragB(bp + (size_t)48 * K);
  const int ksteps = K >> 5;
  for (int s = 0; s < ksteps; ++s) {
    const int k1 = (s + 1) << 5;
    v16bf an  = fragA(ap + k1);
    v16bf bn0 = fragB(bp + k1);
    v16bf bn1 = fragB(bp + (size_t)16 * K + k1);
    v16bf bn2 = fragB(bp + (size_t)32 * K + k1);
    v16bf bn3 = fragB(bp + (size_t)48 * K + k1);
    acc[0] = wmma_bf16(a, b0, acc[0]);
    acc[1] = wmma_bf16(a, b1, acc[1]);
    acc[2] = wmma_bf16(a, b2, acc[2]);
    acc[3] = wmma_bf16(a, b3, acc[3]);
    a = an; b0 = bn0; b1 = bn1; b2 = bn2; b3 = bn3;
  }
  if (Cf) {
#pragma unroll
    for (int j = 0; j < 4; ++j)
#pragma unroll
      for (int i = 0; i < 8; ++i)
        Cf[(size_t)(mt * 16 + rbase + i) * N + ng * 64 + 16 * j + cl] = acc[j][i];
  } else {
#pragma unroll
    for (int j = 0; j < 4; ++j)
#pragma unroll
      for (int i = 0; i < 8; ++i)
        Cb[(size_t)(mt * 16 + rbase + i) * N + ng * 64 + 16 * j + cl] = (bf16)acc[j][i];
  }
}

// ---- host launcher ---------------------------------------------------------

extern "C" void kernel_launch(void* const* d_in, const int* in_sizes, int n_in,
                              void* d_out, int out_size, void* d_ws, size_t ws_size,
                              hipStream_t stream) {
  const int*   ids    = (const int*)  d_in[0];
  const float* emb    = (const float*)d_in[1];
  const float* w_in   = (const float*)d_in[2];
  const float* b_in   = (const float*)d_in[3];
  const float* w_bb   = (const float*)d_in[4];
  const float* b_bb   = (const float*)d_in[5];
  const float* gamma  = (const float*)d_in[6];
  const float* beta   = (const float*)d_in[7];
  const float* w_out  = (const float*)d_in[8];
  const float* w_head = (const float*)d_in[9];
  float* out = (float*)d_out;

  char* ws = (char*)d_ws;
  size_t off = 0;
  auto alloc = [&](size_t bytes) {
    char* p = ws + off; off += (bytes + 255) & ~(size_t)255; return p;
  };
  bf16*  w_inT   = (bf16*) alloc((size_t)NDIM * EDIM * 2);
  bf16*  w_bbT   = (bf16*) alloc((size_t)NDIM * NDIM * 2);
  bf16*  w_outT  = (bf16*) alloc((size_t)EDIM * NDIM * 2);
  bf16*  w_headT = (bf16*) alloc((size_t)VSZ  * EDIM * 2);
  float* x       = (float*)alloc((size_t)MROWS * NDIM * 4);
  bf16*  h       = (bf16*) alloc((size_t)MROWS * NDIM * 2);
  bf16*  mid     = (bf16*) alloc((size_t)MROWS * EDIM * 2);
  bf16*  a_st    = (bf16*) alloc((size_t)16 * NDIM * 2);
  float* s_st    = (float*)alloc((size_t)8 * NDIM * 4);
  unsigned* bar  = (unsigned*)alloc(256);

  hipMemsetAsync(bar, 0, 256, stream);

  auto cdiv = [](long a, long b) { return (int)((a + b - 1) / b); };

  // 1) weights -> bf16, transposed (output-major, K contiguous)
  k_convert_T<<<cdiv((long)EDIM * NDIM, 256), 256, 0, stream>>>(w_in,  w_inT,  EDIM, NDIM);
  k_convert_T<<<cdiv((long)NDIM * NDIM, 256), 256, 0, stream>>>(w_bb,  w_bbT,  NDIM, NDIM);
  k_convert_T<<<cdiv((long)NDIM * EDIM, 256), 256, 0, stream>>>(w_out, w_outT, NDIM, EDIM);
  k_convert_T<<<cdiv((long)EDIM * VSZ,  256), 256, 0, stream>>>(w_head,w_headT,EDIM, VSZ);

  // 2) x = emb[ids] @ w_in + b_in : 128 Mtiles x 16 Ngroups = 2048 waves
  k_embed_proj<<<(MROWS / 16) * (NDIM / 64) / 8, 256, 0, stream>>>(ids, emb, w_inT, b_in, x);

  // 3) cooperative recurrent scan (16 WGs x 4 waves; 130KB LDS/WGP, TDM-staged)
  const int rec_lds = RWAVES * 16 * LDSK * (int)sizeof(bf16);
  hipFuncSetAttribute((const void*)k_recurrent,
                      hipFuncAttributeMaxDynamicSharedMemorySize, rec_lds);
  k_recurrent<<<RWGS, RWAVES * 32, rec_lds, stream>>>(x, w_bbT, b_bb, gamma, beta,
                                                      a_st, s_st, h, bar);

  // 4) mid = h @ w_out : (2048x1024)(1024x768) -> bf16
  k_gemm<<<(MROWS / 16) * (EDIM / 64) / 8, 256, 0, stream>>>(h, w_outT, nullptr, mid,
                                                             MROWS, EDIM, NDIM);
  // 5) logits = mid @ w_head : (2048x768)(768x32000) -> f32 d_out
  k_gemm<<<(MROWS / 16) * (VSZ / 64) / 8, 256, 0, stream>>>(mid, w_headT, out, nullptr,
                                                            MROWS, VSZ, EDIM);
}
